// MonotonicAttention_44006234915292
// MI455X (gfx1250) — matrix operand order repaired and checked by reading, not compiled, not run on hardware
//
#include <hip/hip_runtime.h>
#include <hip/hip_bf16.h>
#include <math.h>

typedef __attribute__((ext_vector_type(16))) __bf16 v16bf;
typedef __attribute__((ext_vector_type(8)))  float  v8f;

#define EPS_F 1e-6f
#define T_FULL 1024
#define S_FULL 1024
#define Dh 64
#define PB 1028   // padded pbuf row stride (floats)

// Dynamic LDS layout:
//   vlds : __bf16[1024*64]  = 131072 B   (V staged as bf16, [s][d])
//   pbuf : float[16*PB]     =  65792 B   (p / fixed-alpha block, 16 t-rows)
//   wp   : float[96]        =    384 B   (3x32 wave-partial scratch)
//   prev : float[1024]      =   4096 B   (carry row: unfixed alpha_{t-1})
//   accb : float[16*64]     =   4096 B   (attn block accumulator)
#define SMEM_BYTES (131072 + (16*PB + 96 + 1024 + 1024)*4)

__device__ __forceinline__ float wave_scan_mul(float x, int lane) {
    #pragma unroll
    for (int d = 1; d < 32; d <<= 1) {
        float y = __shfl_up(x, d, 32);
        if (lane >= d) x *= y;
    }
    return x;
}
__device__ __forceinline__ float wave_scan_add(float x, int lane) {
    #pragma unroll
    for (int d = 1; d < 32; d <<= 1) {
        float y = __shfl_up(x, d, 32);
        if (lane >= d) x += y;
    }
    return x;
}
__device__ __forceinline__ float wave_reduce_add(float x) {
    #pragma unroll
    for (int d = 16; d >= 1; d >>= 1) x += __shfl_xor(x, d, 32);
    return x;
}

__global__ void __launch_bounds__(1024)
mono_attn_head_kernel(const float* __restrict__ q, const float* __restrict__ kp,
                      const float* __restrict__ v, const float* __restrict__ ebias,
                      float* __restrict__ alpha_out, float* __restrict__ attn_ws)
{
    extern __shared__ char smem[];
    __bf16* vlds = (__bf16*)smem;
    float*  pbuf = (float*)(smem + 131072);
    float*  wpA  = pbuf + 16*PB;       // 32 floats
    float*  wpB  = wpA + 32;           // 32 floats
    float*  wpC  = wpB + 32;           // 32 floats
    float*  prev = wpC + 32;           // 1024 floats
    float*  accb = prev + 1024;        // 16*64 floats

    const int b    = blockIdx.x;
    const int tid  = threadIdx.x;
    const int lane = tid & 31;
    const int wv   = tid >> 5;           // 0..31
    const int m    = lane & 15;
    const bool hi  = (lane >= 16);
    const float bias = ebias[0];

    const float* qh = q  + (size_t)b * T_FULL * Dh;
    const float* kh = kp + (size_t)b * S_FULL * Dh;
    const float* vh = v  + (size_t)b * S_FULL * Dh;
    float* aoutp = alpha_out + (size_t)b * T_FULL * S_FULL;
    float* wsh   = attn_ws   + (size_t)b * T_FULL * Dh;

    // stage V into LDS as bf16 [s][d]
    for (int i = tid; i < S_FULL*Dh; i += 1024)
        vlds[i] = (__bf16)vh[i];

    // init_attn = e0
    prev[tid] = (tid == 0) ? 1.0f : 0.0f;
    __syncthreads();

    for (int bt = 0; bt < T_FULL/16; ++bt) {
        const int t0 = bt * 16;

        // ---------------- energy = q@k^T + bias ; p = sigmoid ----------------
        // A(q) tile is shared by both of this wave's s-tiles: load once.
        v16bf a0, a1;
        {
            const float* qr = qh + (size_t)(t0 + m)*Dh + (hi ? 8 : 0);
            #pragma unroll
            for (int j = 0; j < 8; ++j) {
                a0[j] = (__bf16)qr[j];      a0[8+j] = (__bf16)qr[16+j];
                a1[j] = (__bf16)qr[32+j];   a1[8+j] = (__bf16)qr[48+j];
            }
        }
        #pragma unroll
        for (int sti = 0; sti < 2; ++sti) {
            const int s0 = (wv*2 + sti) * 16;
            const float* kr = kh + (size_t)(s0 + m)*Dh + (hi ? 8 : 0);
            v16bf b0, b1;
            #pragma unroll
            for (int j = 0; j < 16; ++j) { b0[j] = (__bf16)kr[j + (j & 8)]; b1[j] = (__bf16)kr[32 + j + (j & 8)]; }
            v8f c = {};
            c = __builtin_amdgcn_wmma_f32_16x16x32_bf16(false, a0, false, b0, (short)0, c, false, false);
            c = __builtin_amdgcn_wmma_f32_16x16x32_bf16(false, a1, false, b1, (short)0, c, false, false);
            const int rbase = hi ? 8 : 0;
            #pragma unroll
            for (int i = 0; i < 8; ++i) {
                float e = c[i] + bias;
                pbuf[(rbase + i)*PB + s0 + m] = 1.0f / (1.0f + __expf(-e));
            }
        }
        // prefetch next block's q rows while the recurrence runs
        if (bt + 1 < T_FULL/16 && wv == 0)
            __builtin_prefetch(qh + (size_t)(t0 + 16 + m)*Dh, 0, 1);
        __syncthreads();

        // ---------------- sequential expected-alignment recurrence ----------------
        const int s = tid;   // each thread owns one source position
        for (int tt = 0; tt < 16; ++tt) {
            float p   = pbuf[tt*PB + s];
            float omp = fminf(fmaxf(1.0f - p, EPS_F), 1.0f);

            // ---- exclusive block cumprod of omp (shuffle scan, 2 barriers) ----
            float incl = wave_scan_mul(omp, lane);
            if (lane == 31) wpA[wv] = incl;
            __syncthreads();
            if (wv == 0) wpA[lane] = wave_scan_mul(wpA[lane], lane);
            __syncthreads();
            float woffp = (wv == 0) ? 1.0f : wpA[wv - 1];
            float shp   = __shfl_up(incl, 1, 32);
            float cp    = ((lane == 0) ? 1.0f : shp) * woffp;   // exclusive cumprod
            float cpc   = fmaxf(cp, EPS_F);
            float u     = prev[s] / cpc;

            // ---- inclusive block cumsum of u (2 barriers) ----
            float cs = wave_scan_add(u, lane);
            if (lane == 31) wpB[wv] = cs;
            __syncthreads();
            if (wv == 0) wpB[lane] = wave_scan_add(wpB[lane], lane);
            __syncthreads();
            if (wv > 0) cs += wpB[wv - 1];

            float alpha = fminf(fmaxf(p * cp * cs, 0.0f), 1.0f);
            prev[s] = alpha;                            // UNFIXED carry for t+1

            // ---- residual mass -> last source token (2 barriers) ----
            float rs = (s < 1023) ? alpha : 0.0f;
            rs = wave_reduce_add(rs);
            if (lane == 0) wpC[wv] = rs;
            __syncthreads();
            if (wv == 0) {
                float r = wave_reduce_add(wpC[lane]);
                if (lane == 0) wpC[0] = r;
            }
            __syncthreads();
            float total = wpC[0];
            float afix  = (s == 1023) ? (1.0f - fminf(fmaxf(total, 0.0f), 1.0f)) : alpha;

            aoutp[(size_t)(t0 + tt)*S_FULL + s] = afix; // alpha output
            pbuf[tt*PB + s] = afix;                     // beta for attn GEMM
        }
        __syncthreads();   // pbuf rows complete before attn GEMM reads them

        // ---------------- attn block = beta_block(16x1024) @ V(1024x64) ----------------
        accb[tid] = 0.0f;
        __syncthreads();
        {
            const int n0 = (wv & 3) * 16;   // d-tile
            const int kc = wv >> 2;         // K-chunk 0..7 (128 K each)
            v8f c = {};
            #pragma unroll
            for (int jj = 0; jj < 4; ++jj) {
                const int kbase = kc*128 + jj*32;
                v16bf a;
                const int koffA = hi ? 8 : 0;
                #pragma unroll
                for (int j = 0; j < 8; ++j) {
                    a[j]   = (__bf16)pbuf[m*PB + kbase + koffA + j];
                    a[8+j] = (__bf16)pbuf[m*PB + kbase + 16 + koffA + j];
                }
                v16bf bb;
                const int koffB = hi ? 8 : 0;
                #pragma unroll
                for (int j = 0; j < 16; ++j)
                    bb[j] = vlds[(kbase + koffB + j + (j & 8))*Dh + n0 + m];
                c = __builtin_amdgcn_wmma_f32_16x16x32_bf16(false, a, false, bb,
                                                            (short)0, c, false, false);
            }
            const int rbase = hi ? 8 : 0;
            #pragma unroll
            for (int i = 0; i < 8; ++i)
                atomicAdd(&accb[(rbase + i)*Dh + n0 + m], c[i]);
        }
        __syncthreads();
        wsh[(size_t)(t0 + (tid >> 6))*Dh + (tid & 63)] = accb[tid];
        __syncthreads();   // pbuf/accb reused next block
    }
}

// out[r=(t*4+bi)][eo] = sum_e attn[t,bi,e] * W[eo,e] + b[eo];  attn[t,bi,e]=ws[(bi*16+e/64), t, e%64]
__global__ void __launch_bounds__(256)
out_proj_kernel(const float* __restrict__ ws, const float* __restrict__ w,
                const float* __restrict__ bvec, float* __restrict__ out)
{
    const int tid  = threadIdx.x;
    const int lane = tid & 31;
    const int wv   = tid >> 5;                 // 0..7
    const int m    = lane & 15;
    const bool hi  = (lane >= 16);
    const int m0 = blockIdx.x * 16;
    const int n0 = (blockIdx.y * 8 + wv) * 16;

    const int r  = m0 + m;
    const int t  = r >> 2;
    const int bi = r & 3;

    v8f c = {};
    for (int ks = 0; ks < 32; ++ks) {
        const int kbase = ks * 32;
        const int h     = kbase >> 6;                       // head index (constant per step)
        const int dbase = (kbase & 63) + (hi ? 8 : 0);      // dim-within-head base
        const float* ar = ws + ((size_t)(bi*16 + h)*1024 + t)*64 + dbase;
        v16bf a;
        #pragma unroll
        for (int j = 0; j < 8; ++j) { a[j] = (__bf16)ar[j]; a[8+j] = (__bf16)ar[16+j]; }
        v16bf bb;
        const float* wr = w + (size_t)(n0 + m)*1024 + kbase + (hi ? 8 : 0);
        #pragma unroll
        for (int j = 0; j < 16; ++j) bb[j] = (__bf16)wr[j + (j & 8)];
        c = __builtin_amdgcn_wmma_f32_16x16x32_bf16(false, a, false, bb,
                                                    (short)0, c, false, false);
    }
    const int rbase = hi ? 8 : 0;
    #pragma unroll
    for (int i = 0; i < 8; ++i) {
        const int row = m0 + rbase + i;
        const int col = n0 + m;
        out[(size_t)row*1024 + col] = c[i] + bvec[col];
    }
}

extern "C" void kernel_launch(void* const* d_in, const int* in_sizes, int n_in,
                              void* d_out, int out_size, void* d_ws, size_t ws_size,
                              hipStream_t stream)
{
    const float* q  = (const float*)d_in[0];
    const float* k  = (const float*)d_in[1];
    const float* v  = (const float*)d_in[2];
    const float* ow = (const float*)d_in[3];
    const float* ob = (const float*)d_in[4];
    const float* eb = (const float*)d_in[5];

    float* out   = (float*)d_out;
    float* alpha = out + (size_t)1024*4*1024;         // out first, then alpha
    float* ws    = (float*)d_ws;                      // attn [64][1024][64] f32 = 16 MB

    mono_attn_head_kernel<<<64, 1024, SMEM_BYTES, stream>>>(q, k, v, eb, alpha, ws);

    dim3 g2(256, 8);   // 4096/16 M-tiles x (1024/128) N-groups
    out_proj_kernel<<<g2, 256, 0, stream>>>(ws, ow, ob, out);
}
